// NGRU_20057497272913
// MI455X (gfx1250) — matrix operand order repaired and checked by reading, not compiled, not run on hardware
//
#include <hip/hip_runtime.h>
#include <hip/hip_bf16.h>

// NGRU: only layer L-1 contributes to the output (layers are parallel, output = h[-1]).
// H=64, 3H=192, T=48, B=64, N=256 -> 16384 independent GRU sequences.
// Mixed precision: bf16 WMMA inputs, f32 accumulation / gates / hidden state.

typedef __attribute__((ext_vector_type(16))) __bf16 v16bf;
typedef __attribute__((ext_vector_type(8)))  float  v8f;
typedef __attribute__((ext_vector_type(4)))  float  fv4;   // native vector (nontemporal-load ok)

#define HDIM 64
#define GDIM 192      // 3*H
#define TLEN 48
#define NNODE 256
#define LNUM 2
#define WSTR 72       // padded LDS row stride (halves): 144B rows -> 16B aligned, bank-conflict-free

union AFrag { v16bf v; unsigned w[8]; uint4 q[2]; };

// pack two floats to packed bf16 (truncation) with one v_perm_b32
__device__ __forceinline__ unsigned packbf(float lo, float hi) {
    return __builtin_amdgcn_perm(__float_as_uint(hi), __float_as_uint(lo), 0x07060302u);
}
__device__ __forceinline__ unsigned short truncbf(float f) {
    return (unsigned short)(__float_as_uint(f) >> 16);   // -> ds_store_b16_d16_hi
}
__device__ __forceinline__ float sigmoid_(float x) {
    return __builtin_amdgcn_rcpf(1.0f + __expf(-x));
}
__device__ __forceinline__ float tanh_(float x) {
#if __has_builtin(__builtin_amdgcn_tanhf)
    return __builtin_amdgcn_tanhf(x);                    // v_tanh_f32 (1 trans op)
#else
    float e = __expf(2.0f * x);
    return (e - 1.0f) * __builtin_amdgcn_rcpf(e + 1.0f);
#endif
}

// A fragment: 16x32 bf16 (MxK) from row-major [16][WSTR] bf16 LDS tile (used for h).
// ISA layout: lanes hold M=lane&15; K = kb*32 + (lane>=16)*8 + {0..7} and +16.
__device__ __forceinline__ void loadA(const unsigned short* s, int lane, int kb, AFrag& f) {
    int off = (lane & 15) * WSTR + kb * 32 + ((lane >> 4) & 1) * 8;
    f.q[0] = *(const uint4*)(s + off);
    f.q[1] = *(const uint4*)(s + off + 16);
}
// B fragment: 32x16 bf16 (KxN), B[k][n] = W[g0+n][k], W row-major [GDIM][WSTR] in LDS.
__device__ __forceinline__ void loadB(const unsigned short* w, int lane, int g0, int kb, AFrag& f) {
    int off = (g0 + (lane & 15)) * WSTR + kb * 32 + ((lane >> 4) & 1) * 8;
    f.q[0] = *(const uint4*)(w + off);
    f.q[1] = *(const uint4*)(w + off + 16);
}
__device__ __forceinline__ v8f wmma_bf16(const AFrag& a, const AFrag& b, v8f c) {
    return __builtin_amdgcn_wmma_f32_16x16x32_bf16(false, a.v, false, b.v, (short)0, c, false, false);
}

// Per-lane x loads straight into A-fragment element order (no LDS for x).
// Lane owns row m = lane&15; chunks at col kb*32 + s*16 + lh*8 + q*4.
__device__ __forceinline__ void load_x(const float* xrow, fv4* xc) {
#pragma unroll
    for (int kb = 0; kb < 2; ++kb)
#pragma unroll
        for (int s = 0; s < 2; ++s)
#pragma unroll
            for (int q = 0; q < 2; ++q)
                xc[kb * 4 + s * 2 + q] =
                    __builtin_nontemporal_load((const fv4*)(xrow + kb * 32 + s * 16 + q * 4));
}
__device__ __forceinline__ void build_afrag(const fv4* xc, AFrag* ax) {
#pragma unroll
    for (int kb = 0; kb < 2; ++kb)
#pragma unroll
        for (int s = 0; s < 2; ++s) {
            fv4 g0 = xc[kb * 4 + s * 2 + 0];
            fv4 g1 = xc[kb * 4 + s * 2 + 1];
            ax[kb].w[s * 4 + 0] = packbf(g0.x, g0.y);
            ax[kb].w[s * 4 + 1] = packbf(g0.z, g0.w);
            ax[kb].w[s * 4 + 2] = packbf(g1.x, g1.y);
            ax[kb].w[s * 4 + 3] = packbf(g1.z, g1.w);
        }
}

__global__ __launch_bounds__(64) void ngru_kernel(
    const float* __restrict__ x,   // (B,T,N,H)
    const float* __restrict__ Wih, // (L,3H,H)
    const float* __restrict__ Whh, // (L,3H,H)
    const float* __restrict__ bih, // (L,3H)
    const float* __restrict__ bhh, // (L,3H)
    float* __restrict__ out)       // (B,N,H)
{
    __shared__ unsigned short s_wih[GDIM * WSTR];
    __shared__ unsigned short s_whh[GDIM * WSTR];
    __shared__ unsigned short s_h[2][16 * WSTR];   // per-wave h transpose staging

    const int tid  = threadIdx.x;
    const int lane = tid & 31;
    const int wave = tid >> 5;
    const int wg   = blockIdx.x;                   // 512 blocks
    const int b    = wg >> 3;
    const int n0w  = ((wg & 7) << 5) + (wave << 4); // this wave's 16 sequences

    // Stage layer-(L-1) weights as bf16 into LDS (shared by both waves).
    const float* wihL = Wih + (LNUM - 1) * GDIM * HDIM;
    const float* whhL = Whh + (LNUM - 1) * GDIM * HDIM;
    for (int idx = tid; idx < GDIM * HDIM; idx += 64) {
        int g = idx >> 6, k = idx & 63;
        s_wih[g * WSTR + k] = truncbf(wihL[idx]);
        s_whh[g * WSTR + k] = truncbf(whhL[idx]);
    }
    __syncthreads();

    const int c  = lane & 15;              // tile column (C/D layout)
    const int lh = (lane >> 4) & 1;        // lane half -> row offset 0/8
    const int m  = lane & 15;              // A-layout row owned by this lane

    float bgi[12], bgh[12];
#pragma unroll
    for (int j = 0; j < 12; ++j) {
        bgi[j] = bih[(LNUM - 1) * GDIM + j * 16 + c];
        bgh[j] = bhh[(LNUM - 1) * GDIM + j * 16 + c];
    }

    unsigned short* ha = s_h[wave];

    v8f h[4];
#pragma unroll
    for (int j = 0; j < 4; ++j)
#pragma unroll
        for (int e = 0; e < 8; ++e) h[j][e] = 0.0f;

    // Software pipeline: x_t kept in registers, prefetch x_{t+1} during compute.
    const size_t xrow_base = (((size_t)b * TLEN) * NNODE + n0w + m) * HDIM + lh * 8;
    fv4 xc[8];
    load_x(x + xrow_base, xc);

    for (int t = 0; t < TLEN; ++t) {
        // Build x A-fragments for this step (pure VALU: 16 v_perm_b32).
        AFrag ax[2];
        build_afrag(xc, ax);

        // Prefetch next x tile (clamped; overlaps with this step's compute).
        int tn = (t + 1 < TLEN) ? (t + 1) : t;
        load_x(x + xrow_base + (size_t)tn * NNODE * HDIM, xc);

        // Stage h as bf16 (truncated: straight d16_hi stores) and load h A-fragments.
#pragma unroll
        for (int j = 0; j < 4; ++j)
#pragma unroll
            for (int rr = 0; rr < 8; ++rr)
                ha[(rr + 8 * lh) * WSTR + j * 16 + c] = truncbf(h[j][rr]);

        AFrag ah[2];
        loadA(ha, lane, 0, ah[0]);
        loadA(ha, lane, 1, ah[1]);

        v8f rg[4], zg[4];
        // ---- r gates: tiles 0..3 ----
#pragma unroll
        for (int j = 0; j < 4; ++j) {
            v8f gi, gh;
#pragma unroll
            for (int e = 0; e < 8; ++e) { gi[e] = bgi[j]; gh[e] = bgh[j]; }
            AFrag bw;
            loadB(s_wih, lane, j * 16, 0, bw); gi = wmma_bf16(ax[0], bw, gi);
            loadB(s_wih, lane, j * 16, 1, bw); gi = wmma_bf16(ax[1], bw, gi);
            loadB(s_whh, lane, j * 16, 0, bw); gh = wmma_bf16(ah[0], bw, gh);
            loadB(s_whh, lane, j * 16, 1, bw); gh = wmma_bf16(ah[1], bw, gh);
#pragma unroll
            for (int e = 0; e < 8; ++e) rg[j][e] = sigmoid_(gi[e] + gh[e]);
        }
        // ---- z gates: tiles 4..7 ----
#pragma unroll
        for (int j = 4; j < 8; ++j) {
            v8f gi, gh;
#pragma unroll
            for (int e = 0; e < 8; ++e) { gi[e] = bgi[j]; gh[e] = bgh[j]; }
            AFrag bw;
            loadB(s_wih, lane, j * 16, 0, bw); gi = wmma_bf16(ax[0], bw, gi);
            loadB(s_wih, lane, j * 16, 1, bw); gi = wmma_bf16(ax[1], bw, gi);
            loadB(s_whh, lane, j * 16, 0, bw); gh = wmma_bf16(ah[0], bw, gh);
            loadB(s_whh, lane, j * 16, 1, bw); gh = wmma_bf16(ah[1], bw, gh);
#pragma unroll
            for (int e = 0; e < 8; ++e) zg[j - 4][e] = sigmoid_(gi[e] + gh[e]);
        }
        // ---- n gates + state update: tiles 8..11 ----
#pragma unroll
        for (int j = 8; j < 12; ++j) {
            const int hj = j - 8;
            v8f gi, gh;
#pragma unroll
            for (int e = 0; e < 8; ++e) { gi[e] = bgi[j]; gh[e] = bgh[j]; }
            AFrag bw;
            loadB(s_wih, lane, j * 16, 0, bw); gi = wmma_bf16(ax[0], bw, gi);
            loadB(s_wih, lane, j * 16, 1, bw); gi = wmma_bf16(ax[1], bw, gi);
            loadB(s_whh, lane, j * 16, 0, bw); gh = wmma_bf16(ah[0], bw, gh);
            loadB(s_whh, lane, j * 16, 1, bw); gh = wmma_bf16(ah[1], bw, gh);
#pragma unroll
            for (int e = 0; e < 8; ++e) {
                float nn = tanh_(fmaf(rg[hj][e], gh[e], gi[e]));
                // h = n + z*(h - n)
                h[hj][e] = fmaf(zg[hj][e], h[hj][e] - nn, nn);
            }
        }
    }

    // ---- store final h: out[(b*N + n0w + row)*H + col] ----
    float* op = out + ((size_t)b * NNODE + n0w) * HDIM;
#pragma unroll
    for (int j = 0; j < 4; ++j)
#pragma unroll
        for (int rr = 0; rr < 8; ++rr)
            op[(rr + 8 * lh) * HDIM + j * 16 + c] = h[j][rr];
}

extern "C" void kernel_launch(void* const* d_in, const int* in_sizes, int n_in,
                              void* d_out, int out_size, void* d_ws, size_t ws_size,
                              hipStream_t stream) {
    (void)in_sizes; (void)n_in; (void)out_size; (void)d_ws; (void)ws_size;
    const float* x   = (const float*)d_in[0];
    const float* Wih = (const float*)d_in[1];
    const float* Whh = (const float*)d_in[2];
    const float* bih = (const float*)d_in[3];
    const float* bhh = (const float*)d_in[4];
    float* out = (float*)d_out;
    // 512 blocks x 64 threads (2 waves); each wave owns 16 sequences.
    hipLaunchKernelGGL(ngru_kernel, dim3(512), dim3(64), 0, stream,
                       x, Wih, Whh, bih, bhh, out);
}
